// SeqRnnDecoder_70265664963245
// MI455X (gfx1250) — matrix operand order repaired and checked by reading, not compiled, not run on hardware
//
#include <hip/hip_runtime.h>
#include <cstdint>
#include <cstddef>

// ---------------- problem dims ----------------
#define V_ALL 20000
#define E_DIM 512
#define H_DIM 1024
#define L_DIM 256
#define B_SZ  32
#define S_LEN 256
#define NROWS (B_SZ * S_LEN)   // 8192
// adaptive softmax clusters: [30,100) proj256, [100,1000) proj64, [1000,20000) proj16
#define NC0 70
#define NC1 900
#define NC2 19000
#define NC0P 80
#define NC1P 912
#define NC2P 19008

typedef __attribute__((ext_vector_type(16))) _Float16 v16h;
typedef __attribute__((ext_vector_type(8)))  float    v8f;
typedef __attribute__((ext_vector_type(4)))  int      i32x4;

union HFrag { v16h v; _Float16 h[16]; };

// ---- gfx1250 async global<->LDS DMA (guarded: falls back to plain copies) ----
#if defined(__HIP_DEVICE_COMPILE__) && \
    __has_builtin(__builtin_amdgcn_global_load_async_to_lds_b128) && \
    __has_builtin(__builtin_amdgcn_s_wait_asynccnt)
#define HAVE_ASYNC_LOAD 1
#else
#define HAVE_ASYNC_LOAD 0
#endif
#if defined(__HIP_DEVICE_COMPILE__) && \
    __has_builtin(__builtin_amdgcn_global_store_async_from_lds_b128) && \
    __has_builtin(__builtin_amdgcn_s_wait_asynccnt)
#define HAVE_ASYNC_STORE 1
#else
#define HAVE_ASYNC_STORE 0
#endif

#if HAVE_ASYNC_LOAD
// copy 16B global -> LDS via ASYNCcnt DMA (params are AS1/AS3 int4 pointers)
__device__ __forceinline__ void async_g2l16(const _Float16* g, _Float16* l) {
  __builtin_amdgcn_global_load_async_to_lds_b128(
      (__attribute__((address_space(1))) i32x4*)g,
      (__attribute__((address_space(3))) i32x4*)l, 0, 0);
}
#endif
#if HAVE_ASYNC_STORE
// copy 16B LDS -> global via ASYNCcnt DMA
__device__ __forceinline__ void async_l2g16(_Float16* g, const _Float16* l) {
  __builtin_amdgcn_global_store_async_from_lds_b128(
      (__attribute__((address_space(1))) i32x4*)g,
      (__attribute__((address_space(3))) i32x4*)l, 0, 0);
}
#endif
__device__ __forceinline__ void wait_async_all() {
#if HAVE_ASYNC_LOAD || HAVE_ASYNC_STORE
  __builtin_amdgcn_s_wait_asynccnt(0);
#endif
}

// D = A(16x32) x B(32x16) + C, f16 in / f32 acc  (CDNA5 V_WMMA_F32_16X16X32_F16)
__device__ __forceinline__ v8f wmma_f16(v16h a, v16h b, v8f c) {
  return __builtin_amdgcn_wmma_f32_16x16x32_f16(false, a, false, b, (short)0, c,
                                                false, false);
}

// A fragment: rows m0..m0+15, cols k0..k0+31 of row-major [M, ld].
// Layout (16-bit A 16x32, contiguous k to match B): lane<16 holds row (lane), K={0..15};
// lane>=16 holds row (lane-16), K={16..31}.
__device__ __forceinline__ v16h frag_A(const _Float16* A, int ld, int m0, int k0,
                                       int lane) {
  HFrag f;
  const _Float16* p = A + (size_t)(m0 + (lane & 15)) * ld + k0 + ((lane >> 4) << 4);
#pragma unroll
  for (int i = 0; i < 16; ++i) f.h[i] = p[i];
  return f.v;
}

__device__ __forceinline__ v16h frag_A_k(const _Float16* A, int ld, int m0, int k0,
                                         int kval, int lane) {
  HFrag f;
  const int r  = m0 + (lane & 15);
  const int kb = k0 + ((lane >> 4) << 4);
#pragma unroll
  for (int i = 0; i < 16; ++i) {
    const int k1 = kb + i;
    f.h[i] = (k1 < kval) ? A[(size_t)r * ld + k1] : (_Float16)0.f;
  }
  return f.v;
}

// B fragment = W^T tile (K=32 x N=16) from row-major W [N_total, ld=K_total].
// Each lane reads 16 contiguous f16 (32B) -> coalesced b128 pairs.
__device__ __forceinline__ v16h frag_B(const _Float16* W, int ld, int n0, int k0,
                                       int lane) {
  HFrag f;
  const _Float16* p = W + (size_t)(n0 + (lane & 15)) * ld + k0 + ((lane >> 4) << 4);
#pragma unroll
  for (int i = 0; i < 16; ++i) f.h[i] = p[i];
  return f.v;
}

__device__ __forceinline__ v16h frag_B_k(const _Float16* W, int ld, int n0, int k0,
                                         int kval, int lane) {
  HFrag f;
  const int n  = n0 + (lane & 15);
  const int kb = k0 + ((lane >> 4) << 4);
#pragma unroll
  for (int i = 0; i < 16; ++i) {
    const int k = kb + i;
    f.h[i] = (k < kval) ? W[(size_t)n * ld + k] : (_Float16)0.f;
  }
  return f.v;
}

// ---------------- fp32 -> fp16 weight conversion ----------------
__global__ void k_cvt(const float* __restrict__ src, _Float16* __restrict__ dst, int n) {
  int i = blockIdx.x * blockDim.x + threadIdx.x;
  const int st = gridDim.x * blockDim.x;
  for (; i < n; i += st) dst[i] = (_Float16)src[i];
}

// ---------------- stable argsort(-length) of 32 lengths ----------------
__global__ void k_sort(const int* __restrict__ len, int* __restrict__ sorted) {
  const int i = threadIdx.x;
  if (i < B_SZ) {
    const int li = len[i];
    int r = 0;
    for (int j = 0; j < B_SZ; ++j) {
      const int lj = len[j];
      if (lj > li || (lj == li && j < i)) ++r;
    }
    sorted[r] = i;
  }
}

// ---------------- hid[B,2H] = z @ l2h_w^T + b ----------------
__global__ void k_latent(const _Float16* __restrict__ Z, const _Float16* __restrict__ W,
                         const float* __restrict__ bias, float* __restrict__ hid) {
  const int lane = threadIdx.x & 31;
  const int wave = (blockIdx.x * blockDim.x + threadIdx.x) >> 5;
  const int nw   = (gridDim.x * blockDim.x) >> 5;
  const int nt   = (2 * H_DIM) >> 4;   // 128 col tiles
  for (int t = wave; t < 2 * nt; t += nw) {
    const int m0 = (t / nt) << 4;
    const int n0 = (t % nt) << 4;
    v8f c = {};
#pragma unroll
    for (int k0 = 0; k0 < L_DIM; k0 += 32)
      c = wmma_f16(frag_A(Z, L_DIM, m0, k0, lane), frag_B(W, L_DIM, n0, k0, lane), c);
    const int col = n0 + (lane & 15);
    const int rb  = m0 + ((lane >> 4) << 3);
    const float b = bias[col];
#pragma unroll
    for (int r = 0; r < 8; ++r) hid[(size_t)(rb + r) * (2 * H_DIM) + col] = c[r] + b;
  }
}

// ---------------- persistent fused 2-layer tanh RNN ----------------
// One 1024-thread workgroup (32 waves). h1/h2 + double-buffered x_t in LDS
// (192 KB of the WGP's 320 KB). Per wave all 4 output tiles share one M-tile,
// so each A fragment feeds 4 WMMAs. x_{t+1} is staged with async LDS DMA while
// step t computes; h2_t rows are DMA'd LDS->global.
__global__ void __launch_bounds__(1024)
k_rnn(const _Float16* __restrict__ embh, const int* __restrict__ inp,
      const int* __restrict__ sorted_idx, const float* __restrict__ hid,
      const _Float16* __restrict__ wih0, const _Float16* __restrict__ whh0,
      const float* __restrict__ bih0, const float* __restrict__ bhh0,
      const _Float16* __restrict__ wih1, const _Float16* __restrict__ whh1,
      const float* __restrict__ bih1, const float* __restrict__ bhh1,
      _Float16* __restrict__ outh) {
  extern __shared__ _Float16 sm[];
  _Float16* s_h1 = sm;                          // [32][1024]
  _Float16* s_h2 = sm + B_SZ * H_DIM;           // [32][1024]
  _Float16* s_x0 = sm + 2 * B_SZ * H_DIM;       // [32][512] double buffer
  _Float16* s_x1 = s_x0 + B_SZ * E_DIM;
  __shared__ int s_ord[B_SZ];

  const int tid  = threadIdx.x;
  const int lane = tid & 31;
  const int wave = tid >> 5;

  if (tid < B_SZ) s_ord[tid] = sorted_idx[tid];
  // hidden = hid.reshape(2,B,H); flat idx = l*B*H + b*H + h  (hid is [B,2H] row-major)
  for (int i = tid; i < B_SZ * H_DIM; i += 1024) {
    s_h1[i] = (_Float16)hid[i];
    s_h2[i] = (_Float16)hid[B_SZ * H_DIM + i];
  }
  __syncthreads();

  // per-wave geometry: one M-tile, 4 N-tiles spaced 256 apart
  const int m0 = (wave & 1) << 4;
  const int nb = (wave >> 1) << 4;
  const int colb = lane & 15;
  const int rb   = m0 + ((lane >> 4) << 3);

  // stage x_0 (wave == row; 32B per lane)
  {
    const int tok = inp[s_ord[wave] * S_LEN + 0];
    const _Float16* src = embh + (size_t)tok * E_DIM + lane * 16;
    _Float16* dst = s_x0 + wave * E_DIM + lane * 16;
#if HAVE_ASYNC_LOAD
    async_g2l16(src, dst);
    async_g2l16(src + 8, dst + 8);
#else
    *(float4*)dst       = *(const float4*)src;
    *(float4*)(dst + 8) = *(const float4*)(src + 8);
#endif
  }
  wait_async_all();
  __syncthreads();

  float acc[4][8];
  for (int t = 0; t < S_LEN; ++t) {
    _Float16* s_xc = (t & 1) ? s_x1 : s_x0;
    _Float16* s_xn = (t & 1) ? s_x0 : s_x1;

    // kick off x_{t+1} DMA; prefetch x_{t+2} row into L2
    if (t + 1 < S_LEN) {
      const int tok = inp[s_ord[wave] * S_LEN + t + 1];
      const _Float16* src = embh + (size_t)tok * E_DIM + lane * 16;
      _Float16* dst = s_xn + wave * E_DIM + lane * 16;
#if HAVE_ASYNC_LOAD
      async_g2l16(src, dst);
      async_g2l16(src + 8, dst + 8);
#else
      *(float4*)dst       = *(const float4*)src;
      *(float4*)(dst + 8) = *(const float4*)(src + 8);
#endif
      if (lane == 0 && t + 2 < S_LEN) {   // global_prefetch_b8
        const int tok2 = inp[s_ord[wave] * S_LEN + t + 2];
        __builtin_prefetch(embh + (size_t)tok2 * E_DIM, 0, 1);
      }
    }

    // ---- layer 0: h1_t = tanh(x W0ih^T + h1 W0hh^T + b) ----
    {
      v8f c[4] = {};
#pragma unroll 4
      for (int k0 = 0; k0 < E_DIM; k0 += 32) {
        const v16h a = frag_A(s_xc, E_DIM, m0, k0, lane);
#pragma unroll
        for (int i = 0; i < 4; ++i)
          c[i] = wmma_f16(a, frag_B(wih0, E_DIM, nb + (i << 8), k0, lane), c[i]);
      }
#pragma unroll 4
      for (int k0 = 0; k0 < H_DIM; k0 += 32) {
        const v16h a = frag_A(s_h1, H_DIM, m0, k0, lane);
#pragma unroll
        for (int i = 0; i < 4; ++i)
          c[i] = wmma_f16(a, frag_B(whh0, H_DIM, nb + (i << 8), k0, lane), c[i]);
      }
#pragma unroll
      for (int i = 0; i < 4; ++i) {
        const int col = nb + (i << 8) + colb;
        const float bs = bih0[col] + bhh0[col];
#pragma unroll
        for (int r = 0; r < 8; ++r) acc[i][r] = tanhf(c[i][r] + bs);
      }
    }
    __syncthreads();   // all reads of h1_{t-1} done
#pragma unroll
    for (int i = 0; i < 4; ++i) {
      const int col = nb + (i << 8) + colb;
#pragma unroll
      for (int r = 0; r < 8; ++r) s_h1[(rb + r) * H_DIM + col] = (_Float16)acc[i][r];
    }
    __syncthreads();   // h1_t visible

    // ---- layer 1 ----
    {
      v8f c[4] = {};
#pragma unroll 4
      for (int k0 = 0; k0 < H_DIM; k0 += 32) {
        const v16h a = frag_A(s_h1, H_DIM, m0, k0, lane);
#pragma unroll
        for (int i = 0; i < 4; ++i)
          c[i] = wmma_f16(a, frag_B(wih1, H_DIM, nb + (i << 8), k0, lane), c[i]);
      }
#pragma unroll 4
      for (int k0 = 0; k0 < H_DIM; k0 += 32) {
        const v16h a = frag_A(s_h2, H_DIM, m0, k0, lane);
#pragma unroll
        for (int i = 0; i < 4; ++i)
          c[i] = wmma_f16(a, frag_B(whh1, H_DIM, nb + (i << 8), k0, lane), c[i]);
      }
#pragma unroll
      for (int i = 0; i < 4; ++i) {
        const int col = nb + (i << 8) + colb;
        const float bs = bih1[col] + bhh1[col];
#pragma unroll
        for (int r = 0; r < 8; ++r) acc[i][r] = tanhf(c[i][r] + bs);
      }
    }
    __syncthreads();   // all reads of h2_{t-1} done
#pragma unroll
    for (int i = 0; i < 4; ++i) {
      const int col = nb + (i << 8) + colb;
#pragma unroll
      for (int r = 0; r < 8; ++r) {
        const _Float16 hv = (_Float16)acc[i][r];
        s_h2[(rb + r) * H_DIM + col] = hv;
#if !HAVE_ASYNC_STORE
        const int orig = s_ord[rb + r];                 // out = h2[rev]
        outh[((size_t)orig * S_LEN + t) * H_DIM + col] = hv;
#endif
      }
    }
    __syncthreads();   // h2_t complete in LDS
#if HAVE_ASYNC_STORE
    {   // DMA h2_t rows LDS -> global (original batch order); 64B per lane
      const int orig = s_ord[wave];
      _Float16* dst = outh + ((size_t)orig * S_LEN + t) * H_DIM + lane * 32;
      _Float16* srcl = s_h2 + wave * H_DIM + lane * 32;
#pragma unroll
      for (int q = 0; q < 4; ++q) async_l2g16(dst + q * 8, srcl + q * 8);
    }
#endif
    wait_async_all();  // x_{t+1} landed (and h2 DMA drained before next overwrite)
    __syncthreads();
  }
}

// ------- generic WMMA GEMM: C[M,N] = A[M,K] @ W[N,K]^T, NB n-tiles per A load ----
template <int NB>
__global__ void k_gemm(const _Float16* __restrict__ A, const _Float16* __restrict__ W,
                       int M, int N, int K, float* __restrict__ Cf,
                       _Float16* __restrict__ Ch) {
  const int lane = threadIdx.x & 31;
  const int wave = (blockIdx.x * blockDim.x + threadIdx.x) >> 5;
  const int nw   = (gridDim.x * blockDim.x) >> 5;
  const int mt = M >> 4, ntb = (N >> 4) / NB;
  for (int t = wave; t < mt * ntb; t += nw) {
    const int m0  = (t / ntb) << 4;
    const int nb0 = (t % ntb) * (NB << 4);
    v8f c[NB] = {};
#pragma unroll 4
    for (int k0 = 0; k0 < K; k0 += 32) {
      const v16h a = frag_A(A, K, m0, k0, lane);
#pragma unroll
      for (int j = 0; j < NB; ++j)
        c[j] = wmma_f16(a, frag_B(W, K, nb0 + (j << 4), k0, lane), c[j]);
    }
    const int colb = lane & 15;
    const int rbs  = m0 + ((lane >> 4) << 3);
#pragma unroll
    for (int j = 0; j < NB; ++j) {
      const int col = nb0 + (j << 4) + colb;
#pragma unroll
      for (int r = 0; r < 8; ++r) {
        const size_t o = (size_t)(rbs + r) * N + col;
        if (Cf) Cf[o] = c[j][r];
        else    Ch[o] = (_Float16)c[j][r];
      }
    }
  }
}

// ---------------- fused cluster logits + online logsumexp + target grab ---------
// logits[row, n] = proj[row,:P] . Wc[n,:P];  lse over n<Nc; capture logit at
// n == (tgt[row]-lowIdx). Templated on P so the A fragments are preloaded once
// per 16-row tile and held in registers for the whole N sweep.
template <int P>
__global__ void k_cluster(const _Float16* __restrict__ proj,
                          const _Float16* __restrict__ Wc,
                          int Nc, int lowIdx,
                          const int* __restrict__ tgt,
                          float* __restrict__ lse, float* __restrict__ tl) {
  constexpr int NK = (P + 31) / 32;
  const int lane = threadIdx.x & 31;
  const int wave = (blockIdx.x * blockDim.x + threadIdx.x) >> 5;
  const int nw   = (gridDim.x * blockDim.x) >> 5;
  const int mt   = NROWS >> 4;
  const int ntile = (Nc + 15) >> 4;
  for (int m = wave; m < mt; m += nw) {
    const int m0  = m << 4;
    const int col = lane & 15;
    const int rb  = m0 + ((lane >> 4) << 3);
    v16h af[NK];
#pragma unroll
    for (int kk = 0; kk < NK; ++kk)
      af[kk] = (P >= 32) ? frag_A(proj, P, m0, kk * 32, lane)
                         : frag_A_k(proj, P, m0, kk * 32, P, lane);
    float mv[8], sv[8], tv[8];
    int tloc[8];
#pragma unroll
    for (int r = 0; r < 8; ++r) {
      mv[r] = -3.0e38f; sv[r] = 0.f; tv[r] = -3.0e38f;
      tloc[r] = tgt[rb + r] - lowIdx;
    }
    for (int ntl = 0; ntl < ntile; ++ntl) {
      const int n0 = ntl << 4;
      v8f c = {};
#pragma unroll
      for (int kk = 0; kk < NK; ++kk)
        c = wmma_f16(af[kk],
                     (P % 32 == 0) ? frag_B(Wc, P, n0, kk * 32, lane)
                                   : frag_B_k(Wc, P, n0, kk * 32, P, lane),
                     c);
      const int n = n0 + col;
      const bool valid = (n < Nc);
#pragma unroll
      for (int r = 0; r < 8; ++r) {
        const float v = c[r];
        if (valid) {
          if (v > mv[r]) { sv[r] = sv[r] * expf(mv[r] - v) + 1.f; mv[r] = v; }
          else           { sv[r] += expf(v - mv[r]); }
          if (n == tloc[r]) tv[r] = v;
        }
      }
    }
#pragma unroll
    for (int r = 0; r < 8; ++r) {
      float m_ = mv[r], s_ = sv[r], t_ = tv[r];
      for (int off = 1; off < 16; off <<= 1) {
        const float mo = __shfl_xor(m_, off, 16);
        const float so = __shfl_xor(s_, off, 16);
        const float to = __shfl_xor(t_, off, 16);
        const float mn = fmaxf(m_, mo);
        s_ = s_ * expf(m_ - mn) + so * expf(mo - mn);
        m_ = mn;
        t_ = fmaxf(t_, to);
      }
      if ((lane & 15) == 0) {
        const int row = rb + r;
        lse[row] = m_ + logf(s_);
        tl[row]  = t_;
      }
    }
  }
}

// ---------------- head logsumexp + per-row NLL ----------------
__global__ void k_nll(const float* __restrict__ headl,  // [NROWS, 48], 33 valid
                      const int* __restrict__ tgt,
                      const float* __restrict__ lse0, const float* __restrict__ tl0,
                      const float* __restrict__ lse1, const float* __restrict__ tl1,
                      const float* __restrict__ lse2, const float* __restrict__ tl2,
                      float* __restrict__ nll) {
  const int lane = threadIdx.x & 31;
  const int wave = (blockIdx.x * blockDim.x + threadIdx.x) >> 5;
  const int nw   = (gridDim.x * blockDim.x) >> 5;
  for (int row = wave; row < NROWS; row += nw) {
    const float* h = headl + (size_t)row * 48;
    const float v  = h[lane];                         // logits 0..31
    const float v2 = (lane == 0) ? h[32] : -3.0e38f;  // logit 32
    float m_ = fmaxf(v, v2);
    for (int off = 1; off < 32; off <<= 1) m_ = fmaxf(m_, __shfl_xor(m_, off, 32));
    float e = expf(v - m_) + ((lane == 0) ? expf(v2 - m_) : 0.f);
    for (int off = 1; off < 32; off <<= 1) e += __shfl_xor(e, off, 32);
    if (lane == 0) {
      const float lseh = m_ + logf(e);
      const int t = tgt[row];
      float r = 0.f;
      if (t != 0) {
        if (t < 30)        r = -(h[t] - lseh);
        else if (t < 100)  r = -((h[30] - lseh) + (tl0[row] - lse0[row]));
        else if (t < 1000) r = -((h[31] - lseh) + (tl1[row] - lse1[row]));
        else               r = -((h[32] - lseh) + (tl2[row] - lse2[row]));
      }
      nll[row] = r;
    }
  }
}

// ---------------- deterministic final sum ----------------
__global__ void __launch_bounds__(1024)
k_reduce(const float* __restrict__ x, int n, float* __restrict__ out) {
  __shared__ float sh[1024];
  const int tid = threadIdx.x;
  float s = 0.f;
  for (int i = tid; i < n; i += 1024) s += x[i];
  sh[tid] = s;
  __syncthreads();
  for (int w = 512; w > 0; w >>= 1) {
    if (tid < w) sh[tid] += sh[tid + w];
    __syncthreads();
  }
  if (tid == 0) out[0] = sh[0];
}

// =====================================================================
extern "C" void kernel_launch(void* const* d_in, const int* in_sizes, int n_in,
                              void* d_out, int out_size, void* d_ws, size_t ws_size,
                              hipStream_t stream) {
  (void)in_sizes; (void)n_in; (void)out_size; (void)ws_size;
  const float* z    = (const float*)d_in[0];
  const float* emb  = (const float*)d_in[1];
  const float* l2hw = (const float*)d_in[2];
  const float* l2hb = (const float*)d_in[3];
  const float* wih0 = (const float*)d_in[4];
  const float* whh0 = (const float*)d_in[5];
  const float* bih0 = (const float*)d_in[6];
  const float* bhh0 = (const float*)d_in[7];
  const float* wih1 = (const float*)d_in[8];
  const float* whh1 = (const float*)d_in[9];
  const float* bih1 = (const float*)d_in[10];
  const float* bhh1 = (const float*)d_in[11];
  const float* hw   = (const float*)d_in[12];
  const float* tp0  = (const float*)d_in[13];
  const float* tw0  = (const float*)d_in[14];
  const float* tp1  = (const float*)d_in[15];
  const float* tw1  = (const float*)d_in[16];
  const float* tp2  = (const float*)d_in[17];
  const float* tw2  = (const float*)d_in[18];
  const int* inp    = (const int*)d_in[19];
  const int* tgt    = (const int*)d_in[20];
  const int* len    = (const int*)d_in[21];

  // ---- workspace carve-out (256B aligned) ----
  char* w = (char*)d_ws;
  size_t off = 0;
  auto alloc = [&](size_t bytes) -> void* {
    off = (off + 255) & ~(size_t)255;
    void* p = (void*)(w + off);
    off += bytes;
    return p;
  };
  _Float16* embh  = (_Float16*)alloc((size_t)V_ALL * E_DIM * 2);
  _Float16* wih0h = (_Float16*)alloc((size_t)H_DIM * E_DIM * 2);
  _Float16* whh0h = (_Float16*)alloc((size_t)H_DIM * H_DIM * 2);
  _Float16* wih1h = (_Float16*)alloc((size_t)H_DIM * H_DIM * 2);
  _Float16* whh1h = (_Float16*)alloc((size_t)H_DIM * H_DIM * 2);
  _Float16* l2hh  = (_Float16*)alloc((size_t)2 * H_DIM * L_DIM * 2);
  _Float16* zh    = (_Float16*)alloc((size_t)B_SZ * L_DIM * 2);
  _Float16* hwh   = (_Float16*)alloc((size_t)48 * H_DIM * 2);     // padded 33->48 rows
  _Float16* tp0h  = (_Float16*)alloc((size_t)256 * H_DIM * 2);
  _Float16* tw0h  = (_Float16*)alloc((size_t)NC0P * 256 * 2);     // padded rows zeroed
  _Float16* tp1h  = (_Float16*)alloc((size_t)64 * H_DIM * 2);
  _Float16* tw1h  = (_Float16*)alloc((size_t)NC1P * 64 * 2);
  _Float16* tp2h  = (_Float16*)alloc((size_t)16 * H_DIM * 2);
  _Float16* tw2h  = (_Float16*)alloc((size_t)NC2P * 16 * 2);
  float*    hidf  = (float*)alloc((size_t)2 * B_SZ * H_DIM * 4);
  int*      sidx  = (int*)alloc(B_SZ * 4);
  _Float16* outh  = (_Float16*)alloc((size_t)NROWS * H_DIM * 2);
  float*    headl = (float*)alloc((size_t)NROWS * 48 * 4);
  _Float16* pr0   = (_Float16*)alloc((size_t)NROWS * 256 * 2);
  _Float16* pr1   = (_Float16*)alloc((size_t)NROWS * 64 * 2);
  _Float16* pr2   = (_Float16*)alloc((size_t)NROWS * 16 * 2);
  float* lse0 = (float*)alloc(NROWS * 4); float* tl0d = (float*)alloc(NROWS * 4);
  float* lse1 = (float*)alloc(NROWS * 4); float* tl1d = (float*)alloc(NROWS * 4);
  float* lse2 = (float*)alloc(NROWS * 4); float* tl2d = (float*)alloc(NROWS * 4);
  float* nllr = (float*)alloc(NROWS * 4);

  // ---- zero padded weight buffers, then convert fp32 -> fp16 ----
  (void)hipMemsetAsync(hwh,  0, (size_t)48 * H_DIM * 2, stream);
  (void)hipMemsetAsync(tw0h, 0, (size_t)NC0P * 256 * 2, stream);
  (void)hipMemsetAsync(tw1h, 0, (size_t)NC1P * 64 * 2, stream);
  (void)hipMemsetAsync(tw2h, 0, (size_t)NC2P * 16 * 2, stream);

  auto cvt = [&](const float* s, _Float16* d, int n) {
    int blocks = (n + 255) / 256; if (blocks > 1024) blocks = 1024;
    k_cvt<<<blocks, 256, 0, stream>>>(s, d, n);
  };
  cvt(emb,  embh,  V_ALL * E_DIM);
  cvt(wih0, wih0h, H_DIM * E_DIM);
  cvt(whh0, whh0h, H_DIM * H_DIM);
  cvt(wih1, wih1h, H_DIM * H_DIM);
  cvt(whh1, whh1h, H_DIM * H_DIM);
  cvt(l2hw, l2hh,  2 * H_DIM * L_DIM);
  cvt(z,    zh,    B_SZ * L_DIM);
  cvt(hw,   hwh,   33 * H_DIM);
  cvt(tp0,  tp0h,  256 * H_DIM);
  cvt(tw0,  tw0h,  NC0 * 256);
  cvt(tp1,  tp1h,  64 * H_DIM);
  cvt(tw1,  tw1h,  NC1 * 64);
  cvt(tp2,  tp2h,  16 * H_DIM);
  cvt(tw2,  tw2h,  NC2 * 16);

  // ---- stable argsort(-length) ----
  k_sort<<<1, 32, 0, stream>>>(len, sidx);

  // ---- hid = z @ l2h_w^T + b   (256 tiles -> 256 waves) ----
  k_latent<<<32, 256, 0, stream>>>(zh, l2hh, l2hb, hidf);

  // ---- persistent fused RNN (1 WGP, 192 KB LDS) ----
  const size_t rnn_lds = (size_t)(2 * B_SZ * H_DIM + 2 * B_SZ * E_DIM) * 2;  // 196608 B
  k_rnn<<<1, 1024, rnn_lds, stream>>>(embh, inp, sidx, hidf,
                                      wih0h, whh0h, bih0, bhh0,
                                      wih1h, whh1h, bih1, bhh1, outh);

  // ---- head logits [8192,48] (f32) and tail projections (f16) ----
  k_gemm<3><<<64, 256, 0, stream>>>(outh, hwh,  NROWS, 48,  H_DIM, headl, nullptr);
  k_gemm<4><<<256, 256, 0, stream>>>(outh, tp0h, NROWS, 256, H_DIM, nullptr, pr0);
  k_gemm<4><<<64, 256, 0, stream>>>(outh, tp1h, NROWS, 64,  H_DIM, nullptr, pr1);
  k_gemm<1><<<64, 256, 0, stream>>>(outh, tp2h, NROWS, 16,  H_DIM, nullptr, pr2);

  // ---- fused cluster logits + logsumexp + target logit ----
  k_cluster<256><<<64, 256, 0, stream>>>(pr0, tw0h, NC0, 30,   tgt, lse0, tl0d);
  k_cluster<64><<<64, 256, 0, stream>>>(pr1, tw1h, NC1, 100,  tgt, lse1, tl1d);
  k_cluster<16><<<128, 256, 0, stream>>>(pr2, tw2h, NC2, 1000, tgt, lse2, tl2d);

  // ---- per-row NLL, then deterministic tree-sum into d_out ----
  k_nll<<<256, 1024, 0, stream>>>(headl, tgt, lse0, tl0d, lse1, tl1d, lse2, tl2d, nllr);
  k_reduce<<<1, 1024, 0, stream>>>(nllr, NROWS, (float*)d_out);
}
